// PeerAttention_20942260535933
// MI455X (gfx1250) — matrix operand order repaired and checked by reading, not compiled, not run on hardware
//
#include <hip/hip_runtime.h>
#include <hip/hip_bf16.h>

// MI455X / gfx1250: wave32, WMMA 16x16x32 bf16 (f32 accum), async global->LDS staging

typedef __attribute__((ext_vector_type(16))) __bf16 v16bf;
typedef __attribute__((ext_vector_type(2)))  __bf16 bf16x2;
typedef __attribute__((ext_vector_type(8)))  float  v8f;
typedef __attribute__((ext_vector_type(4)))  int    v4i;

#define B_SZ 512
#define L_SZ 128
#define D_SZ 512
#define H_SZ 8
#define HD_SZ 64

#if defined(__has_builtin)
#  if __has_builtin(__builtin_amdgcn_global_load_async_to_lds_b128)
#    define HAVE_ASYNC 1
#  endif
#endif
#ifndef HAVE_ASYNC
#  define HAVE_ASYNC 0
#endif

__device__ __forceinline__ void async_wait0() {
#if HAVE_ASYNC
#  if __has_builtin(__builtin_amdgcn_s_wait_asynccnt)
    __builtin_amdgcn_s_wait_asynccnt(0);
#  else
    asm volatile("s_wait_asynccnt 0x0" ::: "memory");
#  endif
#endif
}

// native f32 -> bf16 (RNE) via hardware cvt
__device__ __forceinline__ unsigned short bf1(float a) {
    __bf16 h = (__bf16)a;
    return __builtin_bit_cast(unsigned short, h);
}
__device__ __forceinline__ unsigned int packbf(float a, float b) {
    bf16x2 p;
    p.x = (__bf16)a;
    p.y = (__bf16)b;
    return __builtin_bit_cast(unsigned int, p);
}

union AFrag { v16bf f; uint4 q[2]; };

// A-operand 16x32 bf16 tile from LDS (row-major, stride ld shorts).
// Lanes 0-15: row M=lane, K {0..7,16..23}; lanes 16-31: row M=lane-16, K {8..15,24..31}.
__device__ __forceinline__ v16bf ldsFragA(const unsigned short* s, int ld, int row0, int k0) {
    int lane = threadIdx.x & 31;
    int r = lane & 15, hi = lane >> 4;
    const unsigned short* p = s + (row0 + r) * ld + k0;
    AFrag fr;
    fr.q[0] = *(const uint4*)(p + hi * 8);
    fr.q[1] = *(const uint4*)(p + 16 + hi * 8);
    return fr.f;
}

// B-operand 32x16 bf16 tile; LDS holds B "column-major as rows": row c = column c, contiguous in K.
// Lanes 0-15: col N=lane, K {0..7,16..23}; lanes 16-31: col N=lane-16, K {8..15,24..31}.
__device__ __forceinline__ v16bf ldsFragB(const unsigned short* s, int ld, int col0, int k0) {
    int lane = threadIdx.x & 31;
    int c = lane & 15, hi = lane >> 4;
    const unsigned short* p = s + (col0 + c) * ld + k0;
    AFrag fr;
    fr.q[0] = *(const uint4*)(p + hi * 8);
    fr.q[1] = *(const uint4*)(p + 16 + hi * 8);
    return fr.f;
}

__device__ __forceinline__ v8f wmma_bf16(v16bf a, v16bf b, v8f c) {
    return __builtin_amdgcn_wmma_f32_16x16x32_bf16(false, a, false, b, (short)0, c, false, false);
}

// issue one async b128 global->LDS copy (per lane), fallback = plain copy
__device__ __forceinline__ void copy_b128_to_lds(const unsigned short* g, unsigned short* l) {
#if HAVE_ASYNC
    __builtin_amdgcn_global_load_async_to_lds_b128(
        (__attribute__((address_space(1))) v4i*)(v4i*)(unsigned short*)g,
        (__attribute__((address_space(3))) v4i*)(v4i*)l, 0, 0);
#else
    *(uint4*)l = *(const uint4*)g;
#endif
}

// ---------------------------------------------------------------------------
// Kernel 1: qkv = x @ in_proj_w.T + in_proj_b, scattered into per-(b,h) bf16
// blocks q/k/v[(b*8+h)*128 + l][64]. Tile 64(M) x 128(N) per WG, K step 32.
// ---------------------------------------------------------------------------
__global__ __launch_bounds__(256)
void qkv_kernel(const float* __restrict__ x, const float* __restrict__ w,
                const float* __restrict__ bias,
                unsigned short* __restrict__ qws, unsigned short* __restrict__ kws,
                unsigned short* __restrict__ vws) {
    __shared__ unsigned short As[64 * 40];
    __shared__ unsigned short Bs[128 * 40];
    const int tid = threadIdx.x;
    const int wave = tid >> 5, lane = tid & 31;
    const int wm = wave & 1, wn = wave >> 1;
    const int m0 = blockIdx.x * 64;
    const int n0 = blockIdx.y * 128;
    const int ln = lane & 15, hi = lane >> 4;

    v8f acc[2][2];
    acc[0][0] = (v8f){}; acc[0][1] = (v8f){}; acc[1][0] = (v8f){}; acc[1][1] = (v8f){};

    for (int kk = 0; kk < D_SZ; kk += 32) {
        // stage A: 64 rows x 32 k (fp32 -> bf16 via hw cvt)
        for (int i = 0; i < 2; ++i) {
            int idx = i * 256 + tid;             // 512 float4 chunks
            int row = idx >> 3, c4 = idx & 7;
            float4 v = *(const float4*)(x + (size_t)(m0 + row) * D_SZ + kk + c4 * 4);
            uint2 pk;
            pk.x = packbf(v.x, v.y);
            pk.y = packbf(v.z, v.w);
            *(uint2*)&As[row * 40 + c4 * 4] = pk;
        }
        // stage B: 128 rows(n) x 32 k of in_proj_w (row n contiguous in k)
        for (int i = 0; i < 4; ++i) {
            int idx = i * 256 + tid;             // 1024 float4 chunks
            int row = idx >> 3, c4 = idx & 7;
            float4 v = *(const float4*)(w + (size_t)(n0 + row) * D_SZ + kk + c4 * 4);
            uint2 pk;
            pk.x = packbf(v.x, v.y);
            pk.y = packbf(v.z, v.w);
            *(uint2*)&Bs[row * 40 + c4 * 4] = pk;
        }
        __syncthreads();
        v16bf a0 = ldsFragA(As, 40, wm * 32, 0);
        v16bf a1 = ldsFragA(As, 40, wm * 32 + 16, 0);
        v16bf b0 = ldsFragB(Bs, 40, wn * 32, 0);
        v16bf b1 = ldsFragB(Bs, 40, wn * 32 + 16, 0);
        acc[0][0] = wmma_bf16(a0, b0, acc[0][0]);
        acc[0][1] = wmma_bf16(a0, b1, acc[0][1]);
        acc[1][0] = wmma_bf16(a1, b0, acc[1][0]);
        acc[1][1] = wmma_bf16(a1, b1, acc[1][1]);
        __syncthreads();
    }

    // epilogue: scatter to q/k/v per-(b,h) blocks
    for (int ti = 0; ti < 2; ++ti)
        for (int tj = 0; tj < 2; ++tj) {
            int n = n0 + wn * 32 + tj * 16 + ln;     // 0..1535
            int section = n >> 9;                    // 0=q 1=k 2=v (uniform per tile)
            int nn = n & 511;
            int head = nn >> 6, d = nn & 63;
            unsigned short* dst = (section == 0) ? qws : ((section == 1) ? kws : vws);
            float bv = bias[n];
            for (int r = 0; r < 8; ++r) {
                int m = m0 + wm * 32 + ti * 16 + r + hi * 8;
                int bb = m >> 7, l = m & 127;
                size_t off = ((size_t)(bb * H_SZ + head) * L_SZ + l) * HD_SZ + d;
                dst[off] = bf1(acc[ti][tj][r] + bv);
            }
        }
}

// ---------------------------------------------------------------------------
// Kernel 2: fused attention per (b,h). 8 waves, wave w owns query rows 16w..16w+15.
// S = (Q K^T)*scale + bias(mask), softmax in registers, O = P V, O overwrites qws.
// ---------------------------------------------------------------------------
__global__ __launch_bounds__(256)
void attn_kernel(unsigned short* __restrict__ qws, const unsigned short* __restrict__ kws,
                 const unsigned short* __restrict__ vws, const float* __restrict__ corr,
                 const int* __restrict__ pmask, const float* __restrict__ bscale) {
    __shared__ unsigned short SM[128 * 72 + 128 * 72 + 64 * 136]; // Q | K | Vt (54272 B)
    __shared__ float biasRow[128];
    unsigned short* Qs = SM;            // 128 x (64, ld 72)
    unsigned short* Ks = SM + 128 * 72; // 128 x (64, ld 72)
    unsigned short* Vt = SM + 256 * 72; // 64  x (128, ld 136), V transposed
    unsigned short* Ps = SM;            // P 128 x (128, ld 136) aliases dead Q|K

    const int tid = threadIdx.x;
    const int bh = blockIdx.x;
    const int b = bh >> 3;
    const size_t base = (size_t)bh * (L_SZ * HD_SZ);

    // stage Q, K (contiguous 128x64 bf16) as async 16B chunks (no VGPR round trip)
    for (int i = 0; i < 4; ++i) {
        int idx = i * 256 + tid;        // 1024 chunks of 8 shorts
        int row = idx >> 3, c = idx & 7;
        copy_b128_to_lds(qws + base + idx * 8, &Qs[row * 72 + c * 8]);
        copy_b128_to_lds(kws + base + idx * 8, &Ks[row * 72 + c * 8]);
    }
    // stage V transposed: Vt[d][l]
    for (int i = tid; i < L_SZ * HD_SZ; i += 256) {
        unsigned short u = vws[base + i];
        int l = i >> 6, d = i & 63;
        Vt[d * 136 + l] = u;
    }
    float bsc = bscale[0];
    for (int n = tid; n < L_SZ; n += 256)
        biasRow[n] = pmask[b * L_SZ + n] ? -1.0e30f : bsc * corr[b * L_SZ + n];
    async_wait0();
    __syncthreads();

    const int wave = tid >> 5, lane = tid & 31;
    const int ln = lane & 15, hi = lane >> 4;
    const int row0 = wave * 16;

    // S = Q K^T : 8 column tiles x 2 k-steps
    v8f S[8];
    for (int t = 0; t < 8; ++t) S[t] = (v8f){};
    for (int kk = 0; kk < HD_SZ; kk += 32) {
        v16bf a = ldsFragA(Qs, 72, row0, kk);
        for (int t = 0; t < 8; ++t) {
            v16bf bb = ldsFragB(Ks, 72, t * 16, kk);
            S[t] = wmma_bf16(a, bb, S[t]);
        }
    }

    // scale + bias, row softmax. Row m = row0 + r + 8*hi lives in one 16-lane half.
    const float scale = 0.125f;  // HD^-0.5
    for (int t = 0; t < 8; ++t) {
        float bv = biasRow[t * 16 + ln];
        for (int r = 0; r < 8; ++r) S[t][r] = S[t][r] * scale + bv;
    }
    float rinv[8];
    for (int r = 0; r < 8; ++r) {
        float mx = -3.0e38f;
        for (int t = 0; t < 8; ++t) mx = fmaxf(mx, S[t][r]);
        for (int m = 1; m < 16; m <<= 1) mx = fmaxf(mx, __shfl_xor(mx, m, 32));
        float sum = 0.0f;
        for (int t = 0; t < 8; ++t) { float p = __expf(S[t][r] - mx); S[t][r] = p; sum += p; }
        for (int m = 1; m < 16; m <<= 1) sum += __shfl_xor(sum, m, 32);
        rinv[r] = 1.0f / sum;
    }

    __syncthreads();  // all waves done reading Q/K before P overwrites them

    // write P (bf16) rows owned by this wave
    for (int t = 0; t < 8; ++t)
        for (int r = 0; r < 8; ++r) {
            int m = row0 + r + hi * 8;
            Ps[m * 136 + t * 16 + ln] = bf1(S[t][r]);
        }

    // O = P V : 4 column tiles x 4 k-steps (wave-local P rows, no barrier needed)
    v8f O[4];
    for (int t = 0; t < 4; ++t) O[t] = (v8f){};
    for (int kk = 0; kk < L_SZ; kk += 32) {
        v16bf a = ldsFragA(Ps, 136, row0, kk);
        for (int t = 0; t < 4; ++t) {
            v16bf bb = ldsFragB(Vt, 136, t * 16, kk);
            O[t] = wmma_bf16(a, bb, O[t]);
        }
    }
    // normalize and overwrite q workspace with O (same (b,h) block; q already consumed)
    for (int t = 0; t < 4; ++t) {
        int n = t * 16 + ln;
        for (int r = 0; r < 8; ++r) {
            int m = row0 + r + hi * 8;
            qws[base + m * HD_SZ + n] = bf1(O[t][r] * rinv[r]);
        }
    }
}

// ---------------------------------------------------------------------------
// Kernel 3: h = x + O @ out_proj_w.T + out_proj_b (fp32 to d_out).
// O is bf16 in (b,h,l,d) layout; a 32-wide k-slice stays inside one head.
// ---------------------------------------------------------------------------
__global__ __launch_bounds__(256)
void oproj_kernel(const unsigned short* __restrict__ ows, const float* __restrict__ w,
                  const float* __restrict__ bias, const float* __restrict__ x,
                  float* __restrict__ out) {
    __shared__ unsigned short As[64 * 40];
    __shared__ unsigned short Bs[128 * 40];
    const int tid = threadIdx.x;
    const int wave = tid >> 5, lane = tid & 31;
    const int wm = wave & 1, wn = wave >> 1;
    const int m0 = blockIdx.x * 64;
    const int n0 = blockIdx.y * 128;
    const int ln = lane & 15, hi = lane >> 4;

    v8f acc[2][2];
    acc[0][0] = (v8f){}; acc[0][1] = (v8f){}; acc[1][0] = (v8f){}; acc[1][1] = (v8f){};

    for (int kk = 0; kk < D_SZ; kk += 32) {
        int head = kk >> 6, d0 = kk & 63;
        {   // stage A: 64 rows x 32 k from (b,h,l,d) bf16 layout, 256 async 16B chunks
            int row = tid >> 2, c = tid & 3;
            int m = m0 + row, bb = m >> 7, l = m & 127;
            size_t off = ((size_t)(bb * H_SZ + head) * L_SZ + l) * HD_SZ + d0 + c * 8;
            copy_b128_to_lds(ows + off, &As[row * 40 + c * 8]);
        }
        for (int i = 0; i < 4; ++i) {   // stage B: out_proj_w rows n, contiguous k
            int idx = i * 256 + tid;
            int row = idx >> 3, c4 = idx & 7;
            float4 v = *(const float4*)(w + (size_t)(n0 + row) * D_SZ + kk + c4 * 4);
            uint2 pk;
            pk.x = packbf(v.x, v.y);
            pk.y = packbf(v.z, v.w);
            *(uint2*)&Bs[row * 40 + c4 * 4] = pk;
        }
        async_wait0();
        __syncthreads();
        v16bf a0 = ldsFragA(As, 40, wm * 32, 0);
        v16bf a1 = ldsFragA(As, 40, wm * 32 + 16, 0);
        v16bf b0 = ldsFragB(Bs, 40, wn * 32, 0);
        v16bf b1 = ldsFragB(Bs, 40, wn * 32 + 16, 0);
        acc[0][0] = wmma_bf16(a0, b0, acc[0][0]);
        acc[0][1] = wmma_bf16(a0, b1, acc[0][1]);
        acc[1][0] = wmma_bf16(a1, b0, acc[1][0]);
        acc[1][1] = wmma_bf16(a1, b1, acc[1][1]);
        __syncthreads();
    }

    for (int ti = 0; ti < 2; ++ti)
        for (int tj = 0; tj < 2; ++tj) {
            int n = n0 + wn * 32 + tj * 16 + ln;
            float bv = bias[n];
            for (int r = 0; r < 8; ++r) {
                int m = m0 + wm * 32 + ti * 16 + r + hi * 8;
                size_t off = (size_t)m * D_SZ + n;
                out[off] = acc[ti][tj][r] + bv + x[off];
            }
        }
}

// ---------------------------------------------------------------------------
// Kernel 4: in-place LayerNorm over rows of 512, one wave32 per row.
// ---------------------------------------------------------------------------
__global__ __launch_bounds__(256)
void ln_kernel(float* __restrict__ out, const float* __restrict__ g,
               const float* __restrict__ be) {
    const int wave = threadIdx.x >> 5, lane = threadIdx.x & 31;
    const size_t row = (size_t)blockIdx.x * 8 + wave;
    float* p = out + row * D_SZ;
    float4 v[4];
    float s = 0.0f, s2 = 0.0f;
    for (int j = 0; j < 4; ++j) {
        v[j] = *(const float4*)(p + j * 128 + lane * 4);
        s  += v[j].x + v[j].y + v[j].z + v[j].w;
        s2 += v[j].x * v[j].x + v[j].y * v[j].y + v[j].z * v[j].z + v[j].w * v[j].w;
    }
    for (int m = 1; m < 32; m <<= 1) {
        s  += __shfl_xor(s,  m, 32);
        s2 += __shfl_xor(s2, m, 32);
    }
    const float mu  = s * (1.0f / D_SZ);
    const float var = s2 * (1.0f / D_SZ) - mu * mu;
    const float inv = rsqrtf(var + 1e-5f);
    for (int j = 0; j < 4; ++j) {
        int c = j * 128 + lane * 4;
        float4 o;
        o.x = (v[j].x - mu) * inv * g[c + 0] + be[c + 0];
        o.y = (v[j].y - mu) * inv * g[c + 1] + be[c + 1];
        o.z = (v[j].z - mu) * inv * g[c + 2] + be[c + 2];
        o.w = (v[j].w - mu) * inv * g[c + 3] + be[c + 3];
        *(float4*)(p + c) = o;
    }
}

// ---------------------------------------------------------------------------
extern "C" void kernel_launch(void* const* d_in, const int* in_sizes, int n_in,
                              void* d_out, int out_size, void* d_ws, size_t ws_size,
                              hipStream_t stream) {
    const float* x      = (const float*)d_in[0];
    const float* corr   = (const float*)d_in[1];
    const int*   pmask  = (const int*)d_in[2];
    const float* in_w   = (const float*)d_in[3];
    const float* in_b   = (const float*)d_in[4];
    const float* out_w  = (const float*)d_in[5];
    const float* out_b  = (const float*)d_in[6];
    const float* ln_w   = (const float*)d_in[7];
    const float* ln_b   = (const float*)d_in[8];
    const float* bscale = (const float*)d_in[9];
    float* out = (float*)d_out;

    const size_t qkvElems = (size_t)B_SZ * L_SZ * D_SZ;   // 33,554,432 bf16 each
    unsigned short* q_ws = (unsigned short*)d_ws;
    unsigned short* k_ws = q_ws + qkvElems;
    unsigned short* v_ws = q_ws + 2 * qkvElems;

    qkv_kernel<<<dim3((B_SZ * L_SZ) / 64, (3 * D_SZ) / 128), 256, 0, stream>>>(
        x, in_w, in_b, q_ws, k_ws, v_ws);
    attn_kernel<<<dim3(B_SZ * H_SZ), 256, 0, stream>>>(
        q_ws, k_ws, v_ws, corr, pmask, bscale);
    oproj_kernel<<<dim3((B_SZ * L_SZ) / 64, D_SZ / 128), 256, 0, stream>>>(
        q_ws, out_w, out_b, x, out);
    ln_kernel<<<dim3((B_SZ * L_SZ) / 8), 256, 0, stream>>>(out, ln_w, ln_b);
}
